// SuperGlue_26182120636643
// MI455X (gfx1250) — matrix operand order repaired and checked by reading, not compiled, not run on hardware
//
#include <hip/hip_runtime.h>
#include <hip/hip_bf16.h>

// ---------------------------------------------------------------------------
// SuperGlue forward for MI455X (gfx1250, wave32).
// All GEMMs run on V_WMMA_F32_16X16X32_F16 (f16 in, f32 accumulate),
// register-blocked 32x32 per wave (2x2 WMMA fragments) for 2x the L2
// arithmetic intensity vs a single-tile microkernel.
// Everything else (BN/LN/softmax/Sinkhorn) stays f32.
// ---------------------------------------------------------------------------

#define NPTS      1024
#define DMODEL    256
#define NHEAD     4
#define HDIM      64
#define SN        1025          // Sinkhorn matrix dimension (with dustbin)
#define SINKIT    100
#define NLAYERS   9
#define THRV      0.2f
#define EPSV      1e-5f

typedef __attribute__((ext_vector_type(16))) _Float16 v16h;
typedef __attribute__((ext_vector_type(8)))  _Float16 v8h;
typedef __attribute__((ext_vector_type(8)))  float    v8f;

// Fragment layouts per CDNA5 ISA 7.12.2 (wave32):
//   A (16x32 f16): lane L holds row M=L%16, elems [0..7]=K(kk+8*(L/16)+0..7),
//                  elems [8..15]=K(kk+16+8*(L/16)+0..7)
//   B (32x16 f16): lane L holds col N=L%16, elems [0..7]=K(kk+8*(L/16)+0..7),
//                  elems [8..15]=K(kk+16+8*(L/16)+0..7)
//                  where B[k][n] = W[tileN*16+n][k]  -> contiguous in W's row
//   C/D (16x16 f32): lane L col N=L%16, VGPR v row M = v + 8*(L/16)

__device__ __forceinline__ v16h frag_a(const _Float16* __restrict__ row, int kk, int half)
{
    const v8h p0 = *(const v8h*)(row + kk + half * 8);
    const v8h p1 = *(const v8h*)(row + kk + 16 + half * 8);
    v16h a;
#pragma unroll
    for (int i = 0; i < 8; ++i) { a[i] = p0[i]; a[i + 8] = p1[i]; }
    return a;
}

__device__ __forceinline__ v16h frag_b(const _Float16* __restrict__ row, int kk, int half)
{
    const v8h p0 = *(const v8h*)(row + kk + half * 8);
    const v8h p1 = *(const v8h*)(row + kk + 16 + half * 8);
    v16h b;
#pragma unroll
    for (int i = 0; i < 8; ++i) { b[i] = p0[i]; b[i + 8] = p1[i]; }
    return b;
}

#define WMMA_F16(A, B, C) \
    __builtin_amdgcn_wmma_f32_16x16x32_f16(false, (A), false, (B), (short)0, (C), false, false)

__device__ __forceinline__ void store_tile(const v8f& acc, int row0, int col,
                                           float alpha, float bv,
                                           float* __restrict__ outF,
                                           _Float16* __restrict__ outH,
                                           int ldc, int half)
{
#pragma unroll
    for (int v = 0; v < 8; ++v) {
        const int   row = row0 + v + half * 8;
        const float val = alpha * acc[v] + bv;
        if (outH) outH[(size_t)row * ldc + col] = (_Float16)val;
        else      outF[(size_t)row * ldc + col] = val;
    }
}

// ---------------------------------------------------------------------------
// Generic WMMA GEMM:  OUT[n][m] = alpha * sum_k X[n][k]*W[m][k]  (+ bias[m])
// X: (N x K) f16 row-major lda; W: (M x K) f16 row-major ldb  (=> X @ W^T)
// OUT: f32 (outF) or f16 (outH), leading dim ldc.
// One wave computes a 32x32 OUT tile (2x2 fragments); 4 waves per block.
// N, M multiples of 32; K multiple of 32.
// ---------------------------------------------------------------------------
__global__ __launch_bounds__(128)
void k_wmma_gemm_nt(const _Float16* __restrict__ X, int lda,
                    const _Float16* __restrict__ W, int ldb,
                    const float* __restrict__ bias, float alpha,
                    float* __restrict__ outF, _Float16* __restrict__ outH,
                    int ldc, int N, int M, int K)
{
    const int wave   = threadIdx.x >> 5;
    const int lane   = threadIdx.x & 31;
    const int tilesN = M >> 5;
    const int nTiles = (N >> 5) * tilesN;
    const int tile   = blockIdx.x * 4 + wave;
    if (tile >= nTiles) return;            // whole-wave exit: EXEC stays full

    const int tm0  = (tile / tilesN) * 32;
    const int tn0  = (tile % tilesN) * 32;
    const int half = lane >> 4;
    const int r16  = lane & 15;

    const _Float16* __restrict__ x0 = X + (size_t)(tm0 + r16) * lda;
    const _Float16* __restrict__ x1 = x0 + (size_t)16 * lda;
    const _Float16* __restrict__ w0 = W + (size_t)(tn0 + r16) * ldb;
    const _Float16* __restrict__ w1 = w0 + (size_t)16 * ldb;

    v8f acc00 = {}, acc01 = {}, acc10 = {}, acc11 = {};

    for (int kk = 0; kk < K; kk += 32) {
        const v16h a0 = frag_a(x0, kk, half);
        const v16h a1 = frag_a(x1, kk, half);
        const v16h b0 = frag_b(w0, kk, half);
        const v16h b1 = frag_b(w1, kk, half);
        acc00 = WMMA_F16(a0, b0, acc00);
        acc01 = WMMA_F16(a0, b1, acc01);
        acc10 = WMMA_F16(a1, b0, acc10);
        acc11 = WMMA_F16(a1, b1, acc11);
    }

    const int   c0  = tn0 + r16;
    const int   c1  = c0 + 16;
    const float bv0 = bias ? bias[c0] : 0.0f;
    const float bv1 = bias ? bias[c1] : 0.0f;
    store_tile(acc00, tm0,      c0, alpha, bv0, outF, outH, ldc, half);
    store_tile(acc01, tm0,      c1, alpha, bv1, outF, outH, ldc, half);
    store_tile(acc10, tm0 + 16, c0, alpha, bv0, outF, outH, ldc, half);
    store_tile(acc11, tm0 + 16, c1, alpha, bv1, outF, outH, ldc, half);
}

// ------------------------- small helper kernels ----------------------------

__global__ void k_normalize(const float* __restrict__ kpts, const int* hp,
                            const int* wp, float* __restrict__ out, int n)
{
    int i = blockIdx.x * blockDim.x + threadIdx.x;
    if (i >= n) return;
    float w = (float)(*wp) - 1.0f, h = (float)(*hp) - 1.0f;
    out[i * 2 + 0] = kpts[i * 2 + 0] / w * 2.0f - 1.0f;
    out[i * 2 + 1] = kpts[i * 2 + 1] / h * 2.0f - 1.0f;
}

__global__ void k_linear(const float* __restrict__ X, const float* __restrict__ W,
                         const float* __restrict__ B, float* __restrict__ Y,
                         int N, int Cin, int Cout)
{
    int idx = blockIdx.x * blockDim.x + threadIdx.x;
    if (idx >= N * Cout) return;
    int n = idx / Cout, co = idx % Cout;
    const float* x = X + (size_t)n * Cin;
    const float* w = W + (size_t)co * Cin;
    float acc = B[co];
    for (int c = 0; c < Cin; ++c) acc += x[c] * w[c];
    Y[idx] = acc;
}

__global__ void k_bn_stats(const float* __restrict__ X, int N, int C,
                           float* __restrict__ mu, float* __restrict__ rstd)
{
    int c = blockIdx.x, t = threadIdx.x;
    __shared__ float s1[256], s2[256];
    float a = 0.f, b = 0.f;
    for (int n = t; n < N; n += 256) { float v = X[(size_t)n * C + c]; a += v; b += v * v; }
    s1[t] = a; s2[t] = b; __syncthreads();
    for (int o = 128; o; o >>= 1) {
        if (t < o) { s1[t] += s1[t + o]; s2[t] += s2[t + o]; }
        __syncthreads();
    }
    if (t == 0) {
        float m = s1[0] / (float)N;
        float var = s2[0] / (float)N - m * m;   // biased variance (train mode)
        mu[c] = m; rstd[c] = rsqrtf(var + EPSV);
    }
}

__global__ void k_bn_relu(float* __restrict__ X, const float* __restrict__ mu,
                          const float* __restrict__ rstd, const float* __restrict__ g,
                          const float* __restrict__ beta, int total, int C)
{
    int idx = blockIdx.x * blockDim.x + threadIdx.x;
    if (idx >= total) return;
    int c = idx % C;
    float v = (X[idx] - mu[c]) * rstd[c] * g[c] + beta[c];
    X[idx] = fmaxf(v, 0.0f);
}

__global__ void k_relu(float* __restrict__ X, int n)
{
    int i = blockIdx.x * blockDim.x + threadIdx.x;
    if (i < n) X[i] = fmaxf(X[i], 0.0f);
}

// F[n][d] = enc[n][d] + desc[d][n]    (desc is (D,N) row-major)
__global__ void k_add_desc(const float* __restrict__ enc, const float* __restrict__ desc,
                           float* __restrict__ F, int N, int D)
{
    int idx = blockIdx.x * blockDim.x + threadIdx.x;
    if (idx >= N * D) return;
    int n = idx / D, d = idx % D;
    F[idx] = enc[idx] + desc[(size_t)d * N + n];
}

__global__ void k_f32_to_f16(const float* __restrict__ s, _Float16* __restrict__ d, int n)
{
    int i = blockIdx.x * blockDim.x + threadIdx.x;
    if (i < n) d[i] = (_Float16)s[i];
}

// dst[c][r] = src[r][c]   src: rows x cols
__global__ void k_transpose_h(const _Float16* __restrict__ src, _Float16* __restrict__ dst,
                              int rows, int cols)
{
    int idx = blockIdx.x * blockDim.x + threadIdx.x;
    if (idx >= rows * cols) return;
    int r = idx / cols, c = idx % cols;
    dst[(size_t)c * rows + r] = src[idx];
}

// Row softmax f32 -> f16 probabilities (block per row)
__global__ void k_softmax_row_h(const float* __restrict__ S, _Float16* __restrict__ A, int cols)
{
    int row = blockIdx.x, t = threadIdx.x;
    const float* s = S + (size_t)row * cols;
    _Float16*    a = A + (size_t)row * cols;
    __shared__ float red[256];
    float mx = -3.4e38f;
    for (int j = t; j < cols; j += 256) mx = fmaxf(mx, s[j]);
    red[t] = mx; __syncthreads();
    for (int o = 128; o; o >>= 1) { if (t < o) red[t] = fmaxf(red[t], red[t + o]); __syncthreads(); }
    mx = red[0]; __syncthreads();
    float sum = 0.f;
    for (int j = t; j < cols; j += 256) sum += expf(s[j] - mx);
    red[t] = sum; __syncthreads();
    for (int o = 128; o; o >>= 1) { if (t < o) red[t] += red[t + o]; __syncthreads(); }
    float inv = 1.0f / red[0];
    for (int j = t; j < cols; j += 256) a[j] = (_Float16)(expf(s[j] - mx) * inv);
}

// out[n][:] = LayerNorm(fin[n] + oproj[n]) * w + b   (blockDim == DMODEL == 256)
__global__ void k_residual_ln(const float* __restrict__ fin, const float* __restrict__ op,
                              const float* __restrict__ w, const float* __restrict__ b,
                              float* __restrict__ fout)
{
    int n = blockIdx.x, t = threadIdx.x;
    float v = fin[(size_t)n * DMODEL + t] + op[(size_t)n * DMODEL + t];
    __shared__ float red[256];
    red[t] = v; __syncthreads();
    for (int o = 128; o; o >>= 1) { if (t < o) red[t] += red[t + o]; __syncthreads(); }
    float mu = red[0] / (float)DMODEL; __syncthreads();
    float d = v - mu;
    red[t] = d * d; __syncthreads();
    for (int o = 128; o; o >>= 1) { if (t < o) red[t] += red[t + o]; __syncthreads(); }
    float rs = rsqrtf(red[0] / (float)DMODEL + EPSV);
    fout[(size_t)n * DMODEL + t] = d * rs * w[t] + b[t];
}

__global__ void k_set_dustbin(float* __restrict__ S)
{
    int i = blockIdx.x * blockDim.x + threadIdx.x;
    if (i >= SN) return;
    S[(size_t)(SN - 1) * SN + i] = THRV;   // last row
    S[(size_t)i * SN + (SN - 1)] = THRV;   // last column
}

// Sinkhorn axis=1: softmax over rows, per column. Block per column.
__global__ void k_sink_col(float* __restrict__ S)
{
    int c = blockIdx.x, t = threadIdx.x;
    __shared__ float red[256];
    float mx = -3.4e38f;
    for (int r = t; r < SN; r += 256) mx = fmaxf(mx, S[(size_t)r * SN + c]);
    red[t] = mx; __syncthreads();
    for (int o = 128; o; o >>= 1) { if (t < o) red[t] = fmaxf(red[t], red[t + o]); __syncthreads(); }
    mx = red[0]; __syncthreads();
    float sum = 0.f;
    for (int r = t; r < SN; r += 256) sum += expf(S[(size_t)r * SN + c] - mx);
    red[t] = sum; __syncthreads();
    for (int o = 128; o; o >>= 1) { if (t < o) red[t] += red[t + o]; __syncthreads(); }
    float inv = 1.0f / red[0];
    for (int r = t; r < SN; r += 256) {
        size_t idx = (size_t)r * SN + c;
        S[idx] = expf(S[idx] - mx) * inv;
    }
}

// Sinkhorn axis=2: softmax over columns, per row. Block per row.
__global__ void k_sink_row(float* __restrict__ S)
{
    int r = blockIdx.x, t = threadIdx.x;
    float* s = S + (size_t)r * SN;
    __shared__ float red[256];
    float mx = -3.4e38f;
    for (int c = t; c < SN; c += 256) mx = fmaxf(mx, s[c]);
    red[t] = mx; __syncthreads();
    for (int o = 128; o; o >>= 1) { if (t < o) red[t] = fmaxf(red[t], red[t + o]); __syncthreads(); }
    mx = red[0]; __syncthreads();
    float sum = 0.f;
    for (int c = t; c < SN; c += 256) sum += expf(s[c] - mx);
    red[t] = sum; __syncthreads();
    for (int o = 128; o; o >>= 1) { if (t < o) red[t] += red[t + o]; __syncthreads(); }
    float inv = 1.0f / red[0];
    for (int c = t; c < SN; c += 256) s[c] = expf(s[c] - mx) * inv;
}

// Per-row max/argmax over the inner 1024x1024 region of S (ld = SN)
__global__ void k_match(const float* __restrict__ S, int* __restrict__ matches,
                        float* __restrict__ scores)
{
    int n = blockIdx.x, t = threadIdx.x;
    const float* s = S + (size_t)n * SN;
    float mx = -3.4e38f; int arg = 0;
    for (int m = t; m < NPTS; m += 256) {
        float v = s[m];
        if (v > mx) { mx = v; arg = m; }
    }
    __shared__ float rv[256]; __shared__ int ra[256];
    rv[t] = mx; ra[t] = arg; __syncthreads();
    for (int o = 128; o; o >>= 1) {
        if (t < o) {
            if (rv[t + o] > rv[t] || (rv[t + o] == rv[t] && ra[t + o] < ra[t])) {
                rv[t] = rv[t + o]; ra[t] = ra[t + o];
            }
        }
        __syncthreads();
    }
    if (t == 0) {
        scores[n]  = rv[0];
        matches[n] = (rv[0] < THRV) ? -1 : ra[0];
    }
}

// ------------------------------ host side ----------------------------------

struct AttnW {
    const _Float16 *Wq, *Wk, *Wv, *Wo;
    const float *bq, *bk, *bv, *bo, *lnw, *lnb;
};

static inline char* bump(char*& p, size_t bytes)
{
    char* r = p;
    p += (bytes + 255) & ~(size_t)255;
    return r;
}

extern "C" void kernel_launch(void* const* d_in, const int* in_sizes, int n_in,
                              void* d_out, int out_size, void* d_ws, size_t ws_size,
                              hipStream_t stream)
{
    (void)in_sizes; (void)n_in; (void)out_size; (void)ws_size;
    auto F = [&](int i) { return (const float*)d_in[i]; };

    // Input order: setup_inputs() dict order, nested `params` flattened per JAX
    // pytree convention (dict keys sorted alphabetically, lists in order).
    const float* kpts0 = F(0);
    const float* kpts1 = F(1);
    const float* desc0 = F(2);
    const float* desc1 = F(3);
    int p = 4;
    // params['cross']: Wk, Wo, Wq, Wv, bk, bo, bq, bv, ln_b, ln_w
    const float* cWk = F(p++); const float* cWo = F(p++);
    const float* cWq = F(p++); const float* cWv = F(p++);
    const float* cbk = F(p++); const float* cbo = F(p++);
    const float* cbq = F(p++); const float* cbv = F(p++);
    const float* clnb = F(p++); const float* clnw = F(p++);
    // params['enc']: W[5], b[5], beta[4], g[4]
    const float* eW[5]; for (int i = 0; i < 5; ++i) eW[i] = F(p++);
    const float* eB[5]; for (int i = 0; i < 5; ++i) eB[i] = F(p++);
    const float* eBe[4]; for (int i = 0; i < 4; ++i) eBe[i] = F(p++);
    const float* eG[4]; for (int i = 0; i < 4; ++i) eG[i] = F(p++);
    // params['self']: same key order as cross
    const float* sWk = F(p++); const float* sWo = F(p++);
    const float* sWq = F(p++); const float* sWv = F(p++);
    const float* sbk = F(p++); const float* sbo = F(p++);
    const float* sbq = F(p++); const float* sbv = F(p++);
    const float* slnb = F(p++); const float* slnw = F(p++);
    const int* h0 = (const int*)d_in[p++]; const int* w0 = (const int*)d_in[p++];
    const int* h1 = (const int*)d_in[p++]; const int* w1 = (const int*)d_in[p++];

    // ---------------- workspace carve-up ----------------
    char* ws = (char*)d_ws;
    const size_t FB = (size_t)NPTS * DMODEL;           // 262144 elements
    float*    F0A = (float*)bump(ws, FB * 4);
    float*    F0B = (float*)bump(ws, FB * 4);
    float*    F1A = (float*)bump(ws, FB * 4);
    float*    F1B = (float*)bump(ws, FB * 4);
    _Float16* H0A = (_Float16*)bump(ws, FB * 2);
    _Float16* H0B = (_Float16*)bump(ws, FB * 2);
    _Float16* H1A = (_Float16*)bump(ws, FB * 2);
    _Float16* H1B = (_Float16*)bump(ws, FB * 2);
    _Float16* Qh  = (_Float16*)bump(ws, FB * 2);
    _Float16* Kh  = (_Float16*)bump(ws, FB * 2);
    _Float16* Vh  = (_Float16*)bump(ws, FB * 2);
    _Float16* VT  = (_Float16*)bump(ws, FB * 2);
    _Float16* Oh  = (_Float16*)bump(ws, FB * 2);
    float*    Oacc  = (float*)bump(ws, FB * 4);
    float*    Oproj = (float*)bump(ws, FB * 4);
    float*    Sc  = (float*)bump(ws, (size_t)NHEAD * NPTS * NPTS * 4);    // 16 MB
    _Float16* Ah  = (_Float16*)bump(ws, (size_t)NHEAD * NPTS * NPTS * 2); // 8 MB
    float*    Sb  = (float*)bump(ws, (size_t)SN * SN * 4);                // 4.2 MB
    float*    EncA = (float*)bump(ws, FB * 4);
    float*    EncB = (float*)bump(ws, FB * 4);
    float*    Knorm = (float*)bump(ws, (size_t)NPTS * 2 * 4);
    float*    Mu   = (float*)bump(ws, 256 * 4);
    float*    Rstd = (float*)bump(ws, 256 * 4);
    _Float16* hW[8];
    for (int i = 0; i < 8; ++i) hW[i] = (_Float16*)bump(ws, (size_t)DMODEL * DMODEL * 2);

    auto cvt = [&](const float* src, _Float16* dst, int n) {
        k_f32_to_f16<<<(n + 255) / 256, 256, 0, stream>>>(src, dst, n);
    };

    // f16 weight mirrors: self {q,k,v,o} then cross {q,k,v,o}
    cvt(sWq, hW[0], DMODEL * DMODEL); cvt(sWk, hW[1], DMODEL * DMODEL);
    cvt(sWv, hW[2], DMODEL * DMODEL); cvt(sWo, hW[3], DMODEL * DMODEL);
    cvt(cWq, hW[4], DMODEL * DMODEL); cvt(cWk, hW[5], DMODEL * DMODEL);
    cvt(cWv, hW[6], DMODEL * DMODEL); cvt(cWo, hW[7], DMODEL * DMODEL);

    AttnW selfW  = { hW[0], hW[1], hW[2], hW[3], sbq, sbk, sbv, sbo, slnw, slnb };
    AttnW crossW = { hW[4], hW[5], hW[6], hW[7], cbq, cbk, cbv, cbo, clnw, clnb };

    auto gemm = [&](const _Float16* X, int lda, const _Float16* W, int ldb,
                    const float* bias, float alpha, float* oF, _Float16* oH,
                    int ldc, int N, int M, int K) {
        int tiles  = (N / 32) * (M / 32);
        int blocks = (tiles + 3) / 4;
        k_wmma_gemm_nt<<<blocks, 128, 0, stream>>>(X, lda, W, ldb, bias, alpha,
                                                   oF, oH, ldc, N, M, K);
    };

    // ---------------- keypoint encoder ----------------
    const int dims[6] = { 2, 32, 64, 128, 256, 256 };
    auto run_enc = [&](const float* kp, const int* hp, const int* wp,
                       const float* desc, float* Fout, _Float16* Hout) {
        k_normalize<<<(NPTS + 255) / 256, 256, 0, stream>>>(kp, hp, wp, Knorm, NPTS);
        const float* cur = Knorm;
        for (int i = 0; i < 5; ++i) {
            float* out = (i % 2 == 0) ? EncA : EncB;
            int Cin = dims[i], Cout = dims[i + 1];
            int total = NPTS * Cout;
            k_linear<<<(total + 255) / 256, 256, 0, stream>>>(cur, eW[i], eB[i],
                                                              out, NPTS, Cin, Cout);
            if (i < 4) {
                k_bn_stats<<<Cout, 256, 0, stream>>>(out, NPTS, Cout, Mu, Rstd);
                k_bn_relu<<<(total + 255) / 256, 256, 0, stream>>>(out, Mu, Rstd,
                                                                   eG[i], eBe[i], total, Cout);
            } else {
                k_relu<<<(total + 255) / 256, 256, 0, stream>>>(out, total);
            }
            cur = out;
        }
        k_add_desc<<<((int)FB + 255) / 256, 256, 0, stream>>>(cur, desc, Fout, NPTS, DMODEL);
        cvt(Fout, Hout, (int)FB);
    };
    run_enc(kpts0, h0, w0, desc0, F0A, H0A);
    run_enc(kpts1, h1, w1, desc1, F1A, H1A);

    // ---------------- one attention block ----------------
    auto run_attn = [&](const float* fq, const _Float16* fqh, const _Float16* fkvh,
                        float* fout, _Float16* fouth, const AttnW& w) {
        gemm(fqh,  DMODEL, w.Wq, DMODEL, w.bq, 1.0f, nullptr, Qh, DMODEL, NPTS, DMODEL, DMODEL);
        gemm(fkvh, DMODEL, w.Wk, DMODEL, w.bk, 1.0f, nullptr, Kh, DMODEL, NPTS, DMODEL, DMODEL);
        gemm(fkvh, DMODEL, w.Wv, DMODEL, w.bv, 1.0f, nullptr, Vh, DMODEL, NPTS, DMODEL, DMODEL);
        k_transpose_h<<<((int)FB + 255) / 256, 256, 0, stream>>>(Vh, VT, NPTS, DMODEL);
        for (int h = 0; h < NHEAD; ++h)
            gemm(Qh + h * HDIM, DMODEL, Kh + h * HDIM, DMODEL, nullptr, 0.125f,
                 Sc + (size_t)h * NPTS * NPTS, nullptr, NPTS, NPTS, NPTS, HDIM);
        k_softmax_row_h<<<NHEAD * NPTS, 256, 0, stream>>>(Sc, Ah, NPTS);
        for (int h = 0; h < NHEAD; ++h)
            gemm(Ah + (size_t)h * NPTS * NPTS, NPTS, VT + (size_t)h * HDIM * NPTS, NPTS,
                 nullptr, 1.0f, Oacc + h * HDIM, nullptr, DMODEL, NPTS, HDIM, NPTS);
        cvt(Oacc, Oh, (int)FB);
        gemm(Oh, DMODEL, w.Wo, DMODEL, w.bo, 1.0f, Oproj, nullptr, DMODEL, NPTS, DMODEL, DMODEL);
        k_residual_ln<<<NPTS, 256, 0, stream>>>(fq, Oproj, w.lnw, w.lnb, fout);
        cvt(fout, fouth, (int)FB);
    };

    // ---------------- 9 GNN layers (shared self/cross weights) -------------
    for (int layer = 0; layer < NLAYERS; ++layer) {
        run_attn(F0A, H0A, H0A, F0B, H0B, selfW);   // f0 self
        run_attn(F1A, H1A, H1A, F1B, H1B, selfW);   // f1 self
        run_attn(F0B, H0B, H1B, F0A, H0A, crossW);  // f0 cross(f1)
        run_attn(F1B, H1B, H0B, F1A, H1A, crossW);  // f1 cross(f0)
    }

    // ---------------- match score matrix + dustbins ----------------
    gemm(H0A, DMODEL, H1A, DMODEL, nullptr, 1.0f / 16.0f, Sb, nullptr, SN,
         NPTS, NPTS, DMODEL);
    k_set_dustbin<<<(SN + 255) / 256, 256, 0, stream>>>(Sb);

    // ---------------- Sinkhorn ----------------
    for (int it = 0; it < SINKIT; ++it) {
        k_sink_col<<<SN, 256, 0, stream>>>(Sb);   // softmax over axis 1 (rows)
        k_sink_row<<<SN, 256, 0, stream>>>(Sb);   // softmax over axis 2 (cols)
    }

    // ---------------- outputs: matches0 (int32) then scores0 (f32) ---------
    k_match<<<NPTS, 256, 0, stream>>>(Sb, (int*)d_out, (float*)d_out + NPTS);
}
